// SimpleSSM_5016521802563
// MI455X (gfx1250) — matrix operand order repaired and checked by reading, not compiled, not run on hardware
//
#include <hip/hip_runtime.h>
#include <hip/hip_bf16.h>

typedef __bf16 bf16;
typedef __attribute__((ext_vector_type(16))) __bf16 v16bf;
typedef __attribute__((ext_vector_type(8)))  __bf16 bf16x8;
typedef __attribute__((ext_vector_type(8)))  float  v8f;

#ifndef USE_ASYNC_COPY
#define USE_ASYNC_COPY 1   // CDNA5 async global->LDS path (ASYNCcnt)
#endif

// ---------------------------------------------------------------------------
// CDNA5 async global -> LDS copy (16B per lane), tracked by ASYNCcnt.
// VDST operand = per-lane LDS byte offset (low 32 bits of a generic LDS ptr).
// ---------------------------------------------------------------------------
__device__ __forceinline__ void async_copy_b128(const void* g, void* lds)
{
#if USE_ASYNC_COPY
    asm volatile("global_load_async_to_lds_b128 %0, %1, off"
                 :: "v"((unsigned)(uintptr_t)lds),
                    "v"((unsigned long long)(uintptr_t)g)
                 : "memory");
#else
    *(bf16x8*)lds = *(const bf16x8*)g;
#endif
}

__device__ __forceinline__ void async_copy_wait()
{
#if USE_ASYNC_COPY
    asm volatile("s_wait_asynccnt 0" ::: "memory");
#endif
}

// Stage a R x 32 bf16 tile (row stride `srcStride` elems) into LDS
// (row stride 40 elems = 20-bank stride, conflict-free). 256 threads.
template <int R>
__device__ __forceinline__ void stage_tile(const bf16* __restrict__ src, int srcStride,
                                           bf16* lds, int tid)
{
#pragma unroll
    for (int c = 0; c < R / 64; ++c) {
        int chunk = tid + c * 256;        // chunk of 8 elems (16B)
        int row   = chunk >> 2;
        int ck    = (chunk & 3) * 8;
        async_copy_b128(src + (size_t)row * srcStride + ck, &lds[row * 40 + ck]);
    }
}

// ---------------------------------------------------------------------------
// Generic f32 -> bf16 conversion
// ---------------------------------------------------------------------------
__global__ __launch_bounds__(256) void f32_to_bf16_kernel(
    const float* __restrict__ src, bf16* __restrict__ dst, int n)
{
    int t = blockIdx.x * blockDim.x + threadIdx.x;
    if (t < n) dst[t] = (bf16)src[t];
}

// Pad x_proj_w (33 x Kd) into (64 x Kd) bf16, zero-filled beyond row 33.
__global__ __launch_bounds__(256) void pad_xproj_kernel(
    const float* __restrict__ src, bf16* __restrict__ dst, int rows_src, int Kd)
{
    int t = blockIdx.x * blockDim.x + threadIdx.x;
    int total = 64 * Kd;
    if (t >= total) return;
    int o = t / Kd, k = t % Kd;
    dst[t] = (o < rows_src) ? (bf16)src[o * Kd + k] : (bf16)0.0f;
}

// negA[i,n] = -exp(A_log[i,n])
__global__ __launch_bounds__(256) void negA_kernel(
    const float* __restrict__ A_log, float* __restrict__ negA, int n)
{
    int t = blockIdx.x * blockDim.x + threadIdx.x;
    if (t < n) negA[t] = -__expf(A_log[t]);
}

// ---------------------------------------------------------------------------
// Tiled WMMA bf16 GEMM:  C[m,o] = sum_k A[m,k] * W[o,k]   (f32 accumulate)
// A: [M, Kd] bf16, W: [Nn, Kd] bf16, C: [M, Nn] f32.  Nn, Kd compile-time.
// Block = 256 threads (8 waves). Block tile 128 x BN (BN = 128, or 64 when
// Nn < 128). Wave tile 32 x (BN/2): 2 x (BN/32) v_wmma_f32_16x16x32_bf16.
// Double-buffered LDS, async global->LDS staging, one barrier per K-step.
// Requires M%128==0, Nn%BN==0, Kd%32==0.
// ---------------------------------------------------------------------------
template <int Nn, int Kd>
__global__ __launch_bounds__(256) void gemm_bf16_wmma(
    const bf16* __restrict__ A, const bf16* __restrict__ W,
    float* __restrict__ C)
{
    constexpr int BN  = (Nn >= 128) ? 128 : 64;  // block tile N
    constexpr int WTN = BN / 2;                  // wave tile N (64 or 32)
    constexpr int TJ  = WTN / 16;                // 16x16 tiles per wave in N (4 or 2)
    constexpr int ATILE = 128 * 40;              // LDS elems per A buffer
    constexpr int BTILE = BN * 40;               // LDS elems per B buffer

    __shared__ bf16 smem[2 * ATILE + 2 * BTILE];
    // buffer addresses computed arithmetically (no LDS pointer tables:
    // addrspacecast constants in static initializers break ld.lld)

    const int tid  = threadIdx.x;
    const int lane = tid & 31;
    const int wave = tid >> 5;
    const int wm   = wave & 3;     // wave M index (0..3)  -> rows wm*32
    const int wn   = wave >> 2;    // wave N index (0..1)  -> cols wn*WTN
    const int half = lane >> 4;    // lane half (0/1)
    const int r    = lane & 15;

    const int m_base = blockIdx.y * 128;
    const int n_base = blockIdx.x * BN;

    const bf16* Abase = A + (size_t)m_base * Kd;
    const bf16* Wbase = W + (size_t)n_base * Kd;

    v8f acc[2][TJ] = {};

    constexpr int nk = Kd / 32;

    // prologue: stage K-tile 0 into buffer 0
    stage_tile<128>(Abase, Kd, smem, tid);
    stage_tile<BN>(Wbase, Kd, smem + 2 * ATILE, tid);

    for (int kt = 0; kt < nk; ++kt) {
        const int buf = kt & 1;

        async_copy_wait();     // this wave's outstanding async stages done
        __syncthreads();       // -> entire tile `buf` visible to all waves

        if (kt + 1 < nk) {     // prefetch next K-tile into the other buffer
            stage_tile<128>(Abase + (kt + 1) * 32, Kd,
                            smem + (buf ^ 1) * ATILE, tid);
            stage_tile<BN>(Wbase + (kt + 1) * 32, Kd,
                           smem + 2 * ATILE + (buf ^ 1) * BTILE, tid);
        }

        // ---- fragments (CDNA5 bf16 A 16x32 / B 32x16 lane layouts) ----
        const bf16* la = smem + buf * ATILE;
        const bf16* lb = smem + 2 * ATILE + buf * BTILE;
        v16bf afrag[2], bfrag[TJ];
#pragma unroll
        for (int ti = 0; ti < 2; ++ti) {
            int arow = wm * 32 + ti * 16 + r;
            bf16x8 lo = *(const bf16x8*)(&la[arow * 40 + half * 8]);       // K = half*8..
            bf16x8 hi = *(const bf16x8*)(&la[arow * 40 + 16 + half * 8]);  // K = 16+half*8..
#pragma unroll
            for (int j = 0; j < 8; ++j) { afrag[ti][j] = lo[j]; afrag[ti][8 + j] = hi[j]; }
        }
#pragma unroll
        for (int tj = 0; tj < TJ; ++tj) {
            int brow = wn * WTN + tj * 16 + r;
            bf16x8 lo = *(const bf16x8*)(&lb[brow * 40 + half * 8]);       // K = half*8..
            bf16x8 hi = *(const bf16x8*)(&lb[brow * 40 + 16 + half * 8]);
#pragma unroll
            for (int j = 0; j < 8; ++j) { bfrag[tj][j] = lo[j]; bfrag[tj][8 + j] = hi[j]; }
        }

#pragma unroll
        for (int ti = 0; ti < 2; ++ti)
#pragma unroll
            for (int tj = 0; tj < TJ; ++tj)
                acc[ti][tj] = __builtin_amdgcn_wmma_f32_16x16x32_bf16(
                    false, afrag[ti], false, bfrag[tj],
                    (short)0, acc[ti][tj], false, false);
    }

    // ---- store: VGPR v of lane (r, half) holds (M = half*8+v, N = r) ----
    float* cptr = C + (size_t)(m_base + wm * 32 + half * 8) * Nn
                    + n_base + wn * WTN + r;
#pragma unroll
    for (int ti = 0; ti < 2; ++ti)
#pragma unroll
        for (int tj = 0; tj < TJ; ++tj)
#pragma unroll
            for (int v = 0; v < 8; ++v)
                cptr[(size_t)(ti * 16 + v) * Nn + tj * 16] = acc[ti][tj][v];
}

// ---------------------------------------------------------------------------
// Depthwise causal conv (K=4) + SiLU, reading xh = xz[..., :I]
// xz: [B*S, 2I] f32. Writes x_conv as f32 and bf16.
// ---------------------------------------------------------------------------
__global__ __launch_bounds__(256) void conv_silu_kernel(
    const float* __restrict__ xz, const float* __restrict__ conv_w,
    float* __restrict__ xconv_f, bf16* __restrict__ xconv_b,
    int S, int I, int total)
{
    int t = blockIdx.x * blockDim.x + threadIdx.x;
    if (t >= total) return;
    int i  = t % I;
    int bs = t / I;
    int s  = bs % S;
    const int O2 = 2 * I;

    float w0 = conv_w[i * 4 + 0], w1 = conv_w[i * 4 + 1];
    float w2 = conv_w[i * 4 + 2], w3 = conv_w[i * 4 + 3];

    size_t base = (size_t)bs * O2 + i;
    float acc = w3 * xz[base];
    if (s >= 1) acc += w2 * xz[base - (size_t)O2];
    if (s >= 2) acc += w1 * xz[base - 2 * (size_t)O2];
    if (s >= 3) acc += w0 * xz[base - 3 * (size_t)O2];

    float v = acc / (1.0f + __expf(-acc));   // silu
    xconv_f[t] = v;
    xconv_b[t] = (bf16)v;
}

// ---------------------------------------------------------------------------
// Pointwise SSM: y = x_conv * (sum_n B*C*exp(delta*negA) + D) * silu(z)
// proj: [B*S, 64] f32 (col0 = delta raw, 1..16 = B, 17..32 = C)
// ---------------------------------------------------------------------------
__global__ __launch_bounds__(256) void ssm_pointwise_kernel(
    const float* __restrict__ proj, const float* __restrict__ xconv,
    const float* __restrict__ xz, const float* __restrict__ negA,
    const float* __restrict__ Dv, bf16* __restrict__ y_b,
    int I, int total)
{
    int t = blockIdx.x * blockDim.x + threadIdx.x;
    if (t >= total) return;
    int i  = t % I;
    int bs = t / I;

    const float* p = proj + (size_t)bs * 64;
    float draw  = p[0];
    float delta = (draw > 20.0f) ? draw : log1pf(__expf(draw));  // softplus

    float acc = 0.0f;
#pragma unroll
    for (int n = 0; n < 16; ++n) {
        float bc = p[1 + n] * p[17 + n];
        acc += bc * __expf(delta * negA[i * 16 + n]);
    }

    float xc = xconv[t];
    float zv = xz[(size_t)bs * (2 * I) + I + i];
    float y  = xc * (acc + Dv[i]);
    y *= zv / (1.0f + __expf(-zv));          // silu(z) gate
    y_b[t] = (bf16)y;
}

// ---------------------------------------------------------------------------
// Host-side launch
// ---------------------------------------------------------------------------
extern "C" void kernel_launch(void* const* d_in, const int* in_sizes, int n_in,
                              void* d_out, int out_size, void* d_ws, size_t ws_size,
                              hipStream_t stream)
{
    (void)in_sizes; (void)n_in; (void)out_size; (void)ws_size;

    const int B = 2, S = 2048, H = 1024;
    const int I = 2 * H;            // 2048
    const int O1 = 2 * I;           // 4096 (in_proj rows)
    const int BS = B * S;           // 4096

    const float* x         = (const float*)d_in[0];
    const float* in_proj_w = (const float*)d_in[1];
    const float* conv_w    = (const float*)d_in[2];
    const float* x_proj_w  = (const float*)d_in[3];
    const float* A_log     = (const float*)d_in[4];
    const float* Dv        = (const float*)d_in[5];
    const float* out_proj_w= (const float*)d_in[6];
    float* out = (float*)d_out;

    char* ws = (char*)d_ws;
    size_t off = 0;
    auto alloc = [&](size_t bytes) { void* p = ws + off; off += (bytes + 255) & ~(size_t)255; return p; };

    bf16*  x_b      = (bf16*) alloc((size_t)BS * H * 2);       // 8 MiB
    bf16*  w_in_b   = (bf16*) alloc((size_t)O1 * H * 2);       // 8 MiB
    bf16*  w_out_b  = (bf16*) alloc((size_t)H * I * 2);        // 4 MiB
    bf16*  xproj_b  = (bf16*) alloc((size_t)64 * I * 2);       // 256 KiB (padded 33->64)
    float* negA     = (float*)alloc((size_t)I * 16 * 4);       // 128 KiB
    float* proj     = (float*)alloc((size_t)BS * 64 * 4);      // 1 MiB
    float* xz       = (float*)alloc((size_t)BS * O1 * 4);      // 64 MiB
    float* xconv_f  = (float*)alloc((size_t)BS * I * 4);       // 32 MiB
    bf16*  xconv_b  = (bf16*) alloc((size_t)BS * I * 2);       // 16 MiB
    bf16*  y_b      = (bf16*) alloc((size_t)BS * I * 2);       // 16 MiB

    auto blocks = [](int n) { return (n + 255) / 256; };

    // 1) precision conversion
    f32_to_bf16_kernel<<<blocks(BS * H), 256, 0, stream>>>(x, x_b, BS * H);
    f32_to_bf16_kernel<<<blocks(O1 * H), 256, 0, stream>>>(in_proj_w, w_in_b, O1 * H);
    f32_to_bf16_kernel<<<blocks(H * I), 256, 0, stream>>>(out_proj_w, w_out_b, H * I);
    pad_xproj_kernel<<<blocks(64 * I), 256, 0, stream>>>(x_proj_w, xproj_b, 33, I);
    negA_kernel<<<blocks(I * 16), 256, 0, stream>>>(A_log, negA, I * 16);

    // 2) GEMM1: xz = x @ in_proj^T   (4096x1024 x 1024x4096)
    gemm_bf16_wmma<4096, 1024><<<dim3(O1 / 128, BS / 128), 256, 0, stream>>>(x_b, w_in_b, xz);

    // 3) depthwise causal conv + silu
    conv_silu_kernel<<<blocks(BS * I), 256, 0, stream>>>(xz, conv_w, xconv_f, xconv_b, S, I, BS * I);

    // 4) GEMM2: proj = x_conv @ x_proj^T (padded to 64 outputs)
    gemm_bf16_wmma<64, 2048><<<dim3(1, BS / 128), 256, 0, stream>>>(xconv_b, xproj_b, proj);

    // 5) pointwise SSM + gates
    ssm_pointwise_kernel<<<blocks(BS * I), 256, 0, stream>>>(proj, xconv_f, xz, negA, Dv, y_b, I, BS * I);

    // 6) GEMM3: out = y @ out_proj^T  (4096x2048 x 2048x1024)
    gemm_bf16_wmma<1024, 2048><<<dim3(H / 128, BS / 128), 256, 0, stream>>>(y_b, w_out_b, out);
}
